// CrossViewGraphMatcher_45672682225862
// MI455X (gfx1250) — compile-verified, matrix operand order checked by reading, not run on hardware
//
#include <hip/hip_runtime.h>

// ---------------------------------------------------------------------------
// CrossViewGraphMatcher for MI455X (gfx1250, wave32, WMMA)
//   B=8, N=2048 (both sides), D=1024, Dm=256
//   Pipeline:
//     prep:      Wt = bf16(W^T), C = 0
//     proj_norm: P = bf16(l2norm(X @ W + b))  (WMMA bf16, fp32 accum)
//     sim:       M0 = Pd @ Ps^T per batch -> d_out (fp32)   (WMMA bf16)
//     5x { rowlse: R = lse_m(M0 - C) ; collse: C = lse_n(M0 - R) }
//     finalize:  a = exp(M0 - R - C) overwrites d_out; partial scores
//     score_red: score[b] = sum(partials)/2048  -> d_out tail
// ---------------------------------------------------------------------------

#define BB   8
#define NN   2048
#define DD   1024
#define DM   256

typedef __attribute__((ext_vector_type(16))) __bf16    v16bf;
typedef __attribute__((ext_vector_type(8)))  float     v8f;
typedef __attribute__((ext_vector_type(4)))  unsigned  u32x4;

union FragBits  { v16bf v; u32x4 u[2]; };
union FragShort { v16bf v; unsigned short s[16]; };

__device__ __forceinline__ unsigned short f2bf(float f) {
    union { float f; unsigned u; } x; x.f = f;
    unsigned u = x.u;
    unsigned r = (u + 0x7FFFu + ((u >> 16) & 1u)) >> 16;   // RNE
    return (unsigned short)r;
}
__device__ __forceinline__ __bf16 bf16_of(float f) {
    union { unsigned short s; __bf16 b; } u; u.s = f2bf(f); return u.b;
}

// 16 bf16 elems of one A/B fragment row: [p, p+8) and [p+16, p+24)
__device__ __forceinline__ v16bf load_bf16_frag(const __bf16* p) {
    FragBits f;
    f.u[0] = *reinterpret_cast<const u32x4*>(p);
    f.u[1] = *reinterpret_cast<const u32x4*>(p + 16);
    return f.v;
}
// same, but converting from fp32 source on the fly
__device__ __forceinline__ v16bf load_f32_frag(const float* p) {
    FragShort f;
#pragma unroll
    for (int i = 0; i < 8; ++i) f.s[i] = f2bf(p[i]);
#pragma unroll
    for (int i = 0; i < 8; ++i) f.s[8 + i] = f2bf(p[16 + i]);
    return f.v;
}

// ------------------------------- prep --------------------------------------
__global__ void __launch_bounds__(256)
prep_kernel(const float* __restrict__ W, __bf16* __restrict__ Wt,
            float* __restrict__ Cv) {
    int idx = blockIdx.x * 256 + threadIdx.x;          // 262144 threads
    if (idx < DD * DM) {
        int k = idx / DM, c = idx % DM;                // W[k][c]
        Wt[(size_t)c * DD + k] = bf16_of(W[idx]);      // Wt[c][k]
    }
    if (idx < BB * NN) Cv[idx] = 0.0f;
}

// --------------------------- projection + l2norm ---------------------------
// block = 128 thr (4 waves), 16 rows x 256 cols per block; wave w -> cols 64w..
__global__ void __launch_bounds__(128)
proj_norm_kernel(const float* __restrict__ X0, const float* __restrict__ X1,
                 const __bf16* __restrict__ Wt, const float* __restrict__ bias,
                 __bf16* __restrict__ P0, __bf16* __restrict__ P1) {
    const float* X = blockIdx.y ? X1 : X0;
    __bf16*      P = blockIdx.y ? P1 : P0;

    const int tid  = threadIdx.x;
    const int wave = tid >> 5;
    const int lane = tid & 31;
    const int h    = lane >> 4;      // K-half selector
    const int r    = lane & 15;      // row within tile (A) / col within tile (B)
    const int row0 = blockIdx.x * 16;
    const int col0 = wave * 64;

    __shared__ float rowsq[16];
    if (tid < 16) rowsq[tid] = 0.0f;

    v8f acc[4] = {};
    const float* arow = X + (size_t)(row0 + r) * DD;

    for (int kk = 0; kk < DD; kk += 32) {
        v16bf a = load_f32_frag(arow + kk + 8 * h);
#pragma unroll
        for (int t = 0; t < 4; ++t) {
            int c = col0 + t * 16 + r;
            v16bf b = load_bf16_frag(Wt + (size_t)c * DD + kk + 8 * h);
            acc[t] = __builtin_amdgcn_wmma_f32_16x16x32_bf16(
                false, a, false, b, (short)0, acc[t], false, false);
        }
    }

    float vals[4][8];
    float psum[8];
#pragma unroll
    for (int j = 0; j < 8; ++j) psum[j] = 0.0f;
#pragma unroll
    for (int t = 0; t < 4; ++t) {
        int c = col0 + t * 16 + r;
        float bc = bias[c];
#pragma unroll
        for (int j = 0; j < 8; ++j) {
            float v = acc[t][j] + bc;
            vals[t][j] = v;
            psum[j] += v * v;
        }
    }
    // reduce across the 16 lanes sharing a K-half (rows j+8h)
#pragma unroll
    for (int off = 8; off >= 1; off >>= 1)
#pragma unroll
        for (int j = 0; j < 8; ++j) psum[j] += __shfl_xor(psum[j], off, 16);

    __syncthreads();                      // rowsq init visible
    if (r == 0) {
#pragma unroll
        for (int j = 0; j < 8; ++j) atomicAdd(&rowsq[j + 8 * h], psum[j]);
    }
    __syncthreads();

#pragma unroll
    for (int t = 0; t < 4; ++t) {
        int c = col0 + t * 16 + r;
#pragma unroll
        for (int j = 0; j < 8; ++j) {
            int rr = j + 8 * h;
            float nrm   = sqrtf(rowsq[rr]);
            float scale = 1.0f / fmaxf(nrm, 1e-12f);
            P[(size_t)(row0 + rr) * DM + c] = bf16_of(vals[t][j] * scale);
        }
    }
}

// ------------------------- similarity GEMM: M0 = Pd Ps^T -------------------
// block = 256 thr (8 waves) -> 128x128 tile; wave: 32 rows x 64 cols
__global__ void __launch_bounds__(256)
sim_kernel(const __bf16* __restrict__ Pd, const __bf16* __restrict__ Ps,
           float* __restrict__ out) {
    const int b    = blockIdx.z;
    const int tid  = threadIdx.x;
    const int wave = tid >> 5;
    const int lane = tid & 31;
    const int h    = lane >> 4;
    const int r    = lane & 15;

    const int rowBase = blockIdx.y * 128 + (wave & 3) * 32;
    const int colBase = blockIdx.x * 128 + (wave >> 2) * 64;

    const __bf16* __restrict__ Db = Pd + (size_t)b * NN * DM;
    const __bf16* __restrict__ Sb = Ps + (size_t)b * NN * DM;
    float* __restrict__ Ob = out + (size_t)b * NN * NN;

    v8f acc[2][4] = {};

    for (int kk = 0; kk < DM; kk += 32) {
        v16bf afr[2];
#pragma unroll
        for (int mt = 0; mt < 2; ++mt)
            afr[mt] = load_bf16_frag(Db + (size_t)(rowBase + mt * 16 + r) * DM + kk + 8 * h);
#pragma unroll
        for (int nt = 0; nt < 4; ++nt) {
            v16bf bfr = load_bf16_frag(Sb + (size_t)(colBase + nt * 16 + r) * DM + kk + 8 * h);
#pragma unroll
            for (int mt = 0; mt < 2; ++mt)
                acc[mt][nt] = __builtin_amdgcn_wmma_f32_16x16x32_bf16(
                    false, afr[mt], false, bfr, (short)0, acc[mt][nt], false, false);
        }
    }

#pragma unroll
    for (int mt = 0; mt < 2; ++mt)
#pragma unroll
        for (int nt = 0; nt < 4; ++nt)
#pragma unroll
            for (int j = 0; j < 8; ++j) {
                int row = rowBase + mt * 16 + j + 8 * h;
                int col = colBase + nt * 16 + r;
                Ob[(size_t)row * NN + col] = acc[mt][nt][j];
            }
}

// -------------------------- online logsumexp helpers -----------------------
__device__ __forceinline__ void lse_push(float& m, float& s, float x) {
    if (x > m) { s = s * __expf(m - x) + 1.0f; m = x; }
    else       { s += __expf(x - m); }
}
__device__ __forceinline__ void lse_merge(float& m, float& s, float m2, float s2) {
    float M = fmaxf(m, m2);
    s = s * __expf(m - M) + s2 * __expf(m2 - M);
    m = M;
}

// R[b,n] = lse over m of (M0[b,n,m] - C[m]);  one block per row
__global__ void __launch_bounds__(256)
rowlse_kernel(const float* __restrict__ M0, const float* __restrict__ Cv,
              float* __restrict__ Rv) {
    const int bi = blockIdx.y, n = blockIdx.x;
    const float* __restrict__ rowp = M0 + ((size_t)bi * NN + n) * NN;
    const float* __restrict__ Cb   = Cv + (size_t)bi * NN;

    float m = -3.0e38f, s = 0.0f;
    for (int i = threadIdx.x; i < NN; i += 256)
        lse_push(m, s, rowp[i] - Cb[i]);

#pragma unroll
    for (int off = 16; off >= 1; off >>= 1) {
        float m2 = __shfl_xor(m, off, 32);
        float s2 = __shfl_xor(s, off, 32);
        lse_merge(m, s, m2, s2);
    }
    __shared__ float sm[8], ss[8];
    if ((threadIdx.x & 31) == 0) { sm[threadIdx.x >> 5] = m; ss[threadIdx.x >> 5] = s; }
    __syncthreads();
    if (threadIdx.x == 0) {
        float M = sm[0], S = ss[0];
        for (int w = 1; w < 8; ++w) lse_merge(M, S, sm[w], ss[w]);
        Rv[(size_t)bi * NN + n] = M + __logf(S);
    }
}

// C[b,m] = lse over n of (M0[b,n,m] - R[n]);  block covers 64 cols, 4 thr/col
__global__ void __launch_bounds__(256)
collse_kernel(const float* __restrict__ M0, const float* __restrict__ Rv,
              float* __restrict__ Cv) {
    const int bi = blockIdx.y;
    const int m0 = blockIdx.x * 64;
    const int c    = threadIdx.x & 63;
    const int rofs = threadIdx.x >> 6;                 // 0..3
    const float* __restrict__ base = M0 + (size_t)bi * NN * NN;
    const float* __restrict__ Rb   = Rv + (size_t)bi * NN;

    float m = -3.0e38f, s = 0.0f;
    for (int n = rofs; n < NN; n += 4)
        lse_push(m, s, base[(size_t)n * NN + m0 + c] - Rb[n]);

    __shared__ float sm[256], ss[256];
    sm[threadIdx.x] = m; ss[threadIdx.x] = s;
    __syncthreads();
    if (threadIdx.x < 64) {
        float M = sm[threadIdx.x], S = ss[threadIdx.x];
#pragma unroll
        for (int k = 1; k < 4; ++k) lse_merge(M, S, sm[threadIdx.x + 64 * k], ss[threadIdx.x + 64 * k]);
        Cv[(size_t)bi * NN + m0 + c] = M + __logf(S);
    }
}

// a = exp(M0 - R - C) overwrites out; per-block partial of sum(a*M0)
__global__ void __launch_bounds__(256)
finalize_kernel(float* __restrict__ out, const float* __restrict__ Rv,
                const float* __restrict__ Cv, float* __restrict__ partial) {
    const int bi = blockIdx.y;
    const float* __restrict__ Rb = Rv + (size_t)bi * NN;
    const float* __restrict__ Cb = Cv + (size_t)bi * NN;
    const size_t batchBase = (size_t)bi * NN * NN;
    const size_t eBase     = (size_t)blockIdx.x * 4096;

    float local = 0.0f;
#pragma unroll
    for (int i = 0; i < 16; ++i) {
        size_t e   = eBase + (size_t)i * 256 + threadIdx.x;
        size_t idx = batchBase + e;
        int n = (int)(e >> 11);
        int m = (int)(e & 2047);
        float m0v = out[idx];
        float a   = __expf(m0v - Rb[n] - Cb[m]);
        out[idx]  = a;
        local += a * m0v;
    }
#pragma unroll
    for (int off = 16; off >= 1; off >>= 1) local += __shfl_xor(local, off, 32);
    __shared__ float wsum[8];
    if ((threadIdx.x & 31) == 0) wsum[threadIdx.x >> 5] = local;
    __syncthreads();
    if (threadIdx.x == 0) {
        float t = 0.0f;
        for (int w = 0; w < 8; ++w) t += wsum[w];
        partial[(size_t)bi * 1024 + blockIdx.x] = t;
    }
}

__global__ void __launch_bounds__(256)
score_reduce_kernel(const float* __restrict__ partial, float* __restrict__ score) {
    const int bi = blockIdx.x;
    float s = 0.0f;
    for (int i = threadIdx.x; i < 1024; i += 256) s += partial[(size_t)bi * 1024 + i];
#pragma unroll
    for (int off = 16; off >= 1; off >>= 1) s += __shfl_xor(s, off, 32);
    __shared__ float wsum[8];
    if ((threadIdx.x & 31) == 0) wsum[threadIdx.x >> 5] = s;
    __syncthreads();
    if (threadIdx.x == 0) {
        float t = 0.0f;
        for (int w = 0; w < 8; ++w) t += wsum[w];
        score[bi] = t * (1.0f / (float)NN);
    }
}

// ------------------------------- launcher ----------------------------------
extern "C" void kernel_launch(void* const* d_in, const int* in_sizes, int n_in,
                              void* d_out, int out_size, void* d_ws, size_t ws_size,
                              hipStream_t stream) {
    const float* drone = (const float*)d_in[0];   // [8,2048,1024]
    const float* sat   = (const float*)d_in[1];   // [8,2048,1024]
    const float* W     = (const float*)d_in[2];   // [1024,256]
    const float* bias  = (const float*)d_in[3];   // [256]

    float* out   = (float*)d_out;                               // assignment [8,2048,2048]
    float* score = out + (size_t)BB * NN * NN;                  // [8]

    // workspace layout
    char* ws = (char*)d_ws;
    __bf16* Wt = (__bf16*)ws;                 ws += (size_t)DM * DD * sizeof(__bf16);
    __bf16* Pd = (__bf16*)ws;                 ws += (size_t)BB * NN * DM * sizeof(__bf16);
    __bf16* Ps = (__bf16*)ws;                 ws += (size_t)BB * NN * DM * sizeof(__bf16);
    float*  Rv = (float*)ws;                  ws += (size_t)BB * NN * sizeof(float);
    float*  Cv = (float*)ws;                  ws += (size_t)BB * NN * sizeof(float);
    float*  partial = (float*)ws;             // [8,1024]

    // 1) W transpose->bf16, zero C
    prep_kernel<<<dim3((DD * DM) / 256), 256, 0, stream>>>(W, Wt, Cv);

    // 2) projection + l2norm for both inputs (y=0 drone, y=1 sat)
    proj_norm_kernel<<<dim3((BB * NN) / 16, 2), 128, 0, stream>>>(
        drone, sat, Wt, bias, Pd, Ps);

    // 3) M0 = Pd @ Ps^T per batch, written to d_out
    sim_kernel<<<dim3(NN / 128, NN / 128, BB), 256, 0, stream>>>(Pd, Ps, out);

    // 4) Sinkhorn via separable potentials: R = lse_m(M0-C); C = lse_n(M0-R)
    for (int it = 0; it < 5; ++it) {
        rowlse_kernel<<<dim3(NN, BB), 256, 0, stream>>>(out, Cv, Rv);
        collse_kernel<<<dim3(NN / 64, BB), 256, 0, stream>>>(out, Rv, Cv);
    }

    // 5) assignment = exp(M0 - R - C) in place, partial scores, then reduce
    finalize_kernel<<<dim3((NN * NN) / 4096, BB), 256, 0, stream>>>(out, Rv, Cv, partial);
    score_reduce_kernel<<<BB, 256, 0, stream>>>(partial, score);
}
